// SpanKL_18657337933899
// MI455X (gfx1250) — matrix operand order repaired and checked by reading, not compiled, not run on hardware
//
#include <hip/hip_runtime.h>

// ---------------------------------------------------------------------------
// Phase 1: link scores via WMMA.
// One wave (32 lanes) computes 16 consecutive scores:
//   s[i] = (1/sqrt(H)) * sum_h S[b,i,h] * E[b,i+1,h]
// by accumulating D = A x B^ + C with v_wmma_f32_16x16x32_f16 over K=H and
// extracting the diagonal of the 16x16 f32 accumulator.
// ---------------------------------------------------------------------------

typedef __attribute__((ext_vector_type(16))) _Float16 v16h;
typedef __attribute__((ext_vector_type(8)))  float    v8f;

__global__ __launch_bounds__(32) void link_score_wmma_kernel(
    const float* __restrict__ S, const float* __restrict__ E,
    float* __restrict__ scores, int B, int L, int H, int nScores /* = L-1 */)
{
    const int nBlk = (nScores + 15) / 16;
    const int blk  = blockIdx.x % nBlk;
    const int b    = blockIdx.x / nBlk;
    const int i0   = blk * 16;
    const int lane = threadIdx.x;          // wave32: 0..31
    const int rc   = lane & 15;            // A row (M) for loading / B col (N)

    // K order used here (contiguous, same for A and B; not the ISA table order):
    //  A f16 16x32: lanes 0-15 hold K=0..15 of row M=lane; lanes 16-31 K=16..31
    //  B f16 32x16: lanes 0-15 hold K=0..15 of col N=lane; lanes 16-31 K=16..31
    const int kbA = (lane < 16) ? 0 : 16;
    const int kbB = (lane < 16) ? 0 : 16;

    const float* arow = S + (size_t)(b * L + i0 + rc) * H;      // i0+rc <= L-1
    int erowi = i0 + rc + 1;                                    // shifted row
    if (erowi > L - 1) erowi = L - 1;                           // clamp (dropped later)
    const float* brow = E + (size_t)(b * L + erowi) * H;

    v8f acc = {};
    for (int kk = 0; kk < H; kk += 32) {
        const float* ap = arow + kk + kbA;
        const float* bp = brow + kk + kbB;
        v16h av, bv;
#pragma unroll
        for (int e = 0; e < 8; ++e) {
            av[e]     = (_Float16)ap[e];        // K = kbA + e
            av[e + 8] = (_Float16)ap[e + 8];    // K = kbA + 8 + e
        }
#pragma unroll
        for (int e = 0; e < 16; ++e)
            bv[e] = (_Float16)bp[e];            // K = kbB + e
        acc = __builtin_amdgcn_wmma_f32_16x16x32_f16(
            /*neg_a=*/false, av, /*neg_b=*/false, bv,
            /*c_mod=*/(short)0, acc, /*reuse_a=*/false, /*reuse_b=*/false);
    }

    // Diagonal of D: M=r @ lane r in acc[r] (lanes 0-7); M=8+r @ lane 24+r in acc[r].
    const int idx = (lane < 16) ? lane : (lane - 24);
    float v = acc[0];
#pragma unroll
    for (int r = 1; r < 8; ++r) v = (idx == r) ? acc[r] : v;

    const bool diagLane = (lane < 8) || (lane >= 24);
    const int  i = i0 + ((lane < 16) ? lane : (lane - 16));
    if (diagLane && i < nScores)
        scores[(size_t)b * nScores + i] = v * rsqrtf((float)H);
}

// ---------------------------------------------------------------------------
// Phase 2: triangular running log-sum-exp + padding, fused.
// out[b,i,j] = logsumexp(s[b, i .. j-1]) for i < j <= L-1, else 0.
// One wave per output row; scores row staged in LDS; 5-step shuffle scan of
// the (max, scaled-sum) monoid per 32-wide tile; NT stores (write-once data).
// ---------------------------------------------------------------------------

#define NEG_INF (-__builtin_huge_valf())

__global__ __launch_bounds__(256) void span_lcse_kernel(
    const float* __restrict__ scores, float* __restrict__ out,
    int B, int L, int n /* = L-1 */)
{
    __shared__ float smem[2048];

    const int rowsPerBlk = 8;                       // 8 waves per block
    const int bpb = (L + rowsPerBlk - 1) / rowsPerBlk;
    const int b      = blockIdx.x / bpb;
    const int i_base = (blockIdx.x % bpb) * rowsPerBlk;

    const float* srow = scores + (size_t)b * n;
    for (int t = threadIdx.x; t < n; t += blockDim.x)
        smem[t] = srow[t];
    __syncthreads();

    const int wid  = threadIdx.x >> 5;
    const int lane = threadIdx.x & 31;
    const int i    = i_base + wid;
    if (i >= L) return;

    float* orow = out + ((size_t)b * L + i) * L;

    // Zero fill: j in [0, i] for normal rows; whole row for i == L-1.
    const int zend = (i < n) ? i : (L - 1);
    for (int j = lane; j <= zend; j += 32)
        __builtin_nontemporal_store(0.0f, orow + j);

    if (i >= n) return;                              // last row: all zeros done

    float Mc = NEG_INF, Tc = 0.0f;                   // carry (running lse state)
    for (int k0 = i; k0 < n; k0 += 32) {
        const int  k     = k0 + lane;
        const bool valid = (k < n);
        float M = valid ? smem[k] : NEG_INF;
        float T = valid ? 1.0f    : 0.0f;

        // Inclusive wave scan of the logsumexp monoid.
#pragma unroll
        for (int d = 1; d < 32; d <<= 1) {
            float Mo = __shfl_up(M, d, 32);
            float To = __shfl_up(T, d, 32);
            if (lane >= d) {
                float Mn = fmaxf(M, Mo);
                T = T * __expf(M - Mn) + To * __expf(Mo - Mn);
                M = Mn;
            }
        }
        // Merge carry from previous tiles.
        float Mf = fmaxf(Mc, M);
        float Tf = Tc * __expf(Mc - Mf) + T * __expf(M - Mf);
        if (valid)
            __builtin_nontemporal_store(__logf(Tf) + Mf, orow + k + 1);
        // New carry = lane 31's merged state (lane 31 is valid whenever a
        // subsequent tile exists).
        Mc = __shfl(Mf, 31, 32);
        Tc = __shfl(Tf, 31, 32);
    }
}

// ---------------------------------------------------------------------------

extern "C" void kernel_launch(void* const* d_in, const int* in_sizes, int n_in,
                              void* d_out, int out_size, void* d_ws, size_t ws_size,
                              hipStream_t stream) {
    const float* S   = (const float*)d_in[0];   // link_start_hidden [B,L,H] f32
    const float* E   = (const float*)d_in[1];   // link_end_hidden   [B,L,H] f32
    float* out       = (float*)d_out;           // [B,L,L,1] f32
    float* scores    = (float*)d_ws;            // [B,L-1]  scratch

    const int H = 768;
    const int L = 2048;
    const int B = in_sizes[0] / (L * H);
    const int n = L - 1;

    const int nBlk = (n + 15) / 16;             // 16 scores per wave
    hipLaunchKernelGGL(link_score_wmma_kernel, dim3(B * nBlk), dim3(32), 0, stream,
                       S, E, scores, B, L, H, n);

    const int bpb = (L + 7) / 8;                // 8 rows per 256-thread block
    hipLaunchKernelGGL(span_lcse_kernel, dim3(B * bpb), dim3(256), 0, stream,
                       scores, out, B, L, n);
}
